// CMMPlus_63728724738821
// MI455X (gfx1250) — compile-verified, matrix operand order checked by reading, not run on hardware
//
#include <hip/hip_runtime.h>
#include <math.h>
#include <stdint.h>

typedef _Float16 h16;
typedef _Float16 v16h __attribute__((ext_vector_type(16)));
typedef float    v8f  __attribute__((ext_vector_type(8)));

// ---------------------------------------------------------------------------
// Packed WMMA fragment layout (16-bit A-matrix layout from CDNA5 ISA 7.12.2):
//   operand [Tiles16 x Ktiles32] blocks; each block = 32 lanes x 16 halves
//   lane = (row&15) + 16*hiL ; element j holds k = (j&7) + (j>=8?16:0) + hiL*8
// A lane's 16 halves are 32 contiguous bytes -> fragment load = 2x b128.
// ---------------------------------------------------------------------------
__device__ __forceinline__ long pk_dst(long m, long k, int nkt)
{
  long mt = m >> 4, kt = k >> 5;
  int kin = (int)(k & 31);
  int hiL = (kin >> 3) & 1;
  int j   = (kin & 7) | ((kin & 16) >> 1);
  int lane = (int)(m & 15) + (hiL << 4);
  return ((mt * nkt + kt) * 32 + lane) * 16 + j;
}

// ---------------------------------------------------------------------------
// LDS-free WMMA GEMM: C[M,N] = A*B (+bias)(+res)(+relu)
// A,B pre-packed in fragment layout. M%64==0, N%64==0, K%32==0.
// 128 threads = 4 waves; each wave owns a 32x32 quadrant of a 64x64 C tile.
// ---------------------------------------------------------------------------
__global__ __launch_bounds__(128) void k_gemm(
    const h16* __restrict__ Apk, const h16* __restrict__ Bpk,
    const float* __restrict__ bias, const h16* __restrict__ res,
    h16* __restrict__ Ch, float* __restrict__ Cf,
    int M, int N, int K, int relu)
{
  const int tid  = threadIdx.x;
  const int lane = tid & 31;
  const int wave = tid >> 5;
  const int wr   = (wave >> 1) * 32;
  const int wc   = (wave & 1) * 32;
  const long tileM = (long)blockIdx.y * 64;
  const long tileN = (long)blockIdx.x * 64;
  const int nkt = K >> 5;
  const int hi  = lane >> 4;
  const int lr  = lane & 15;

  long mt0 = (tileM + wr) >> 4;
  long nt0 = (tileN + wc) >> 4;
  const h16* a0 = Apk + (mt0      * (long)nkt) * 512 + lane * 16;
  const h16* a1 = Apk + ((mt0 + 1) * (long)nkt) * 512 + lane * 16;
  const h16* b0 = Bpk + (nt0      * (long)nkt) * 512 + lane * 16;
  const h16* b1 = Bpk + ((nt0 + 1) * (long)nkt) * 512 + lane * 16;

  v8f acc[2][2] = {};

  for (int kt = 0; kt < nkt; ++kt) {
    v16h A0 = *(const v16h*)a0;
    v16h A1 = *(const v16h*)a1;
    v16h B0 = *(const v16h*)b0;
    v16h B1 = *(const v16h*)b1;
    // prefetch 2 K-tiles ahead (speculative; dropped if past end)
    __builtin_prefetch(a0 + 1024, 0, 0);
    __builtin_prefetch(a1 + 1024, 0, 0);
    __builtin_prefetch(b0 + 1024, 0, 0);
    __builtin_prefetch(b1 + 1024, 0, 0);
    acc[0][0] = __builtin_amdgcn_wmma_f32_16x16x32_f16(false, A0, false, B0, (short)0, acc[0][0], false, false);
    acc[0][1] = __builtin_amdgcn_wmma_f32_16x16x32_f16(false, A0, false, B1, (short)0, acc[0][1], false, false);
    acc[1][0] = __builtin_amdgcn_wmma_f32_16x16x32_f16(false, A1, false, B0, (short)0, acc[1][0], false, false);
    acc[1][1] = __builtin_amdgcn_wmma_f32_16x16x32_f16(false, A1, false, B1, (short)0, acc[1][1], false, false);
    a0 += 512; a1 += 512; b0 += 512; b1 += 512;
  }

  // epilogue: bias / residual / relu / dual-precision store (C row-major)
  #pragma unroll
  for (int i = 0; i < 2; ++i) {
    #pragma unroll
    for (int j = 0; j < 2; ++j) {
      #pragma unroll
      for (int r = 0; r < 8; ++r) {
        long row = tileM + wr + i * 16 + r + 8 * hi;
        long col = tileN + wc + j * 16 + lr;
        float v = acc[i][j][r];
        if (bias) v += bias[col];
        if (res)  v += (float)res[row * (long)N + col];
        if (relu) v = v > 0.f ? v : 0.f;
        long idx = row * (long)N + col;
        if (Ch) Ch[idx] = (h16)v;
        if (Cf) Cf[idx] = v;
      }
    }
  }
}

// ---------------------------------------------------------------------------
// Operand producers (write packed fragment layout directly)
// ---------------------------------------------------------------------------
__global__ void k_pack_a(const h16* __restrict__ src, h16* __restrict__ dst,
                         int M, int K)
{
  long total = (long)M * K;
  long i = blockIdx.x * 256L + threadIdx.x;
  if (i >= total) return;
  long m = i / K, k = i % K;
  dst[pk_dst(m, k, K >> 5)] = src[i];
}

// im2col producing packed A; k-order = (ky*KW+kx)*C + c
__global__ void k_im2col(const h16* __restrict__ src, h16* __restrict__ dst,
                         int Nn, int H, int W, int C, int KH, int KW,
                         int stride, int pad, int OH, int OW, int Kreal, int Kp)
{
  long total = (long)Nn * OH * OW * Kp;
  long i = blockIdx.x * 256L + threadIdx.x;
  if (i >= total) return;
  int nkt = Kp >> 5;
  int j    = i & 15;
  int lane = (i >> 4) & 31;
  long rest = i >> 9;
  int kt = (int)(rest % nkt);
  long mt = rest / nkt;
  long m = mt * 16 + (lane & 15);
  int  k = kt * 32 + (j & 7) + ((j & 8) << 1) + ((lane >> 4) << 3);
  h16 v = (h16)0.f;
  if (k < Kreal) {
    int c = k % C; int t = k / C;
    int kx = t % KW; int ky = t / KW;
    int ox = m % OW; long t2 = m / OW;
    int oy = t2 % OH; int n = (int)(t2 / OH);
    int iy = oy * stride - pad + ky;
    int ix = ox * stride - pad + kx;
    if (iy >= 0 && iy < H && ix >= 0 && ix < W)
      v = src[(((long)n * H + iy) * W + ix) * C + c];
  }
  dst[i] = v;
}

// conv weight OIHW -> packed B [Kp x Co] with BN scale folded
__global__ void k_prep_wconv(const float* __restrict__ w, const float* __restrict__ g,
                             const float* __restrict__ var, h16* __restrict__ Bt,
                             int Co, int Ci, int KH, int KW, int Kp)
{
  long total = (long)Kp * Co;
  long i = blockIdx.x * 256L + threadIdx.x;
  if (i >= total) return;
  int nkt = Kp >> 5;
  int j    = i & 15;
  int lane = (i >> 4) & 31;
  long rest = i >> 9;
  int kt = (int)(rest % nkt);
  long nt = rest / nkt;
  int n = (int)(nt * 16 + (lane & 15));
  int k = kt * 32 + (j & 7) + ((j & 8) << 1) + ((lane >> 4) << 3);
  float val = 0.f;
  int Kreal = Ci * KH * KW;
  if (k < Kreal) {
    int c = k % Ci; int t = k / Ci;
    int kx = t % KW; int ky = t / KW;
    float s = g[n] * rsqrtf(var[n] + 1e-5f);
    val = w[(((long)n * Ci + c) * KH + ky) * KW + kx] * s;
  }
  Bt[i] = (h16)val;
}

__global__ void k_bn_bias(const float* g, const float* b, const float* m,
                          const float* v, float* bias, int Co)
{
  int i = blockIdx.x * 256 + threadIdx.x;
  if (i < Co) { float s = g[i] * rsqrtf(v[i] + 1e-5f); bias[i] = b[i] - m[i] * s; }
}

// linear weight [K,N] f32 -> packed B [Kp x Np] (zero padded)
__global__ void k_prep_lin(const float* __restrict__ W, h16* __restrict__ Bt,
                           int K, int N, int Kp, int Np)
{
  long total = (long)Kp * Np;
  long i = blockIdx.x * 256L + threadIdx.x;
  if (i >= total) return;
  int nkt = Kp >> 5;
  int j    = i & 15;
  int lane = (i >> 4) & 31;
  long rest = i >> 9;
  int kt = (int)(rest % nkt);
  long nt = rest / nkt;
  int n = (int)(nt * 16 + (lane & 15));
  int k = kt * 32 + (j & 7) + ((j & 8) << 1) + ((lane >> 4) << 3);
  Bt[i] = (k < K && n < N) ? (h16)W[(long)k * N + n] : (h16)0.f;
}

__global__ void k_pad_bias(const float* b, float* d, int N, int Np)
{
  int i = blockIdx.x * 256 + threadIdx.x;
  if (i < Np) d[i] = (i < N) ? b[i] : 0.f;
}

// f32 [M,K] -> row-major f16 [M,Kp] zero padded
__global__ void k_pad_a16(const float* __restrict__ s, h16* __restrict__ d,
                          int M, int K, int Kp)
{
  long total = (long)M * Kp;
  long i = blockIdx.x * 256L + threadIdx.x;
  if (i >= total) return;
  int k = i % Kp; long m = i / Kp;
  d[i] = (k < K) ? (h16)s[m * (long)K + k] : (h16)0.f;
}

__global__ void k_fill(float* p, float v, long n)
{
  long i = blockIdx.x * 256L + threadIdx.x;
  if (i < n) p[i] = v;
}

__global__ void k_nchw_to_nhwc(const float* __restrict__ s, h16* __restrict__ d,
                               int Nn, int C, int H, int W)
{
  long total = (long)Nn * C * H * W;
  long i = blockIdx.x * 256L + threadIdx.x;
  if (i >= total) return;
  int c = i % C; long t = i / C;
  int x = t % W; t /= W;
  int y = t % H; int n = t / H;
  d[i] = (h16)s[(((long)n * C + c) * H + y) * W + x];
}

// ---------------------------------------------------------------------------
// ResNet pooling
// ---------------------------------------------------------------------------
__global__ void k_maxpool(const h16* __restrict__ s, h16* __restrict__ d,
                          int Nn, int H, int W, int C, int OH, int OW)
{
  long total = (long)Nn * OH * OW * C;
  long i = blockIdx.x * 256L + threadIdx.x;
  if (i >= total) return;
  int c = i % C; long t = i / C;
  int ox = t % OW; t /= OW;
  int oy = t % OH; int n = (int)(t / OH);
  float m = -3.4e38f;
  for (int ky = 0; ky < 3; ++ky)
    for (int kx = 0; kx < 3; ++kx) {
      int iy = oy * 2 - 1 + ky, ix = ox * 2 - 1 + kx;
      if (iy >= 0 && iy < H && ix >= 0 && ix < W)
        m = fmaxf(m, (float)s[(((long)n * H + iy) * W + ix) * C + c]);
    }
  d[i] = (h16)m;
}

__global__ void k_avgpool(const h16* __restrict__ s, h16* __restrict__ d,
                          int Nn, int H, int W, int C)
{
  long i = blockIdx.x * 256L + threadIdx.x;
  if (i >= (long)Nn * C) return;
  int c = i % C; int n = (int)(i / C);
  float acc = 0.f;
  for (int y = 0; y < H; ++y)
    for (int x = 0; x < W; ++x)
      acc += (float)s[(((long)n * H + y) * W + x) * C + c];
  d[i] = (h16)(acc / (float)(H * W));
}

// ---------------------------------------------------------------------------
// GAT
// ---------------------------------------------------------------------------
__device__ inline void atomicMaxF(float* addr, float val)
{
  if (val >= 0.f) atomicMax((int*)addr, __float_as_int(val));
  else            atomicMin((unsigned int*)addr, __float_as_uint(val));
}

__global__ void k_gat_scores(const float* __restrict__ h, const float* __restrict__ as_,
                             const float* __restrict__ ad_, float* asrc, float* adst,
                             int Nn, int Hh, int Cc)
{
  int i = blockIdx.x * 256 + threadIdx.x;
  if (i >= Nn * Hh) return;
  int hd = i % Hh; int n = i / Hh;
  float s = 0.f, d = 0.f;
  for (int c = 0; c < Cc; ++c) {
    float x = h[((long)n * Hh + hd) * Cc + c];
    s += x * as_[hd * Cc + c];
    d += x * ad_[hd * Cc + c];
  }
  asrc[i] = s; adst[i] = d;
}

__global__ void k_gat_max(const int* __restrict__ ei, const float* asrc,
                          const float* adst, float* mx, int E, int Nn, int Hh)
{
  int Et = E + Nn;
  long i = blockIdx.x * 256L + threadIdx.x;
  if (i >= (long)Et * Hh) return;
  int hd = i % Hh; int e = (int)(i / Hh);
  int s_ = e < E ? ei[e] : e - E;
  int d_ = e < E ? ei[E + e] : e - E;
  float a = asrc[s_ * Hh + hd] + adst[d_ * Hh + hd];
  a = a > 0.f ? a : 0.2f * a;
  atomicMaxF(&mx[d_ * Hh + hd], a);
}

__global__ void k_gat_exp(const int* __restrict__ ei, const float* asrc,
                          const float* adst, const float* mx, float* den,
                          float* eexp, int E, int Nn, int Hh)
{
  int Et = E + Nn;
  long i = blockIdx.x * 256L + threadIdx.x;
  if (i >= (long)Et * Hh) return;
  int hd = i % Hh; int e = (int)(i / Hh);
  int s_ = e < E ? ei[e] : e - E;
  int d_ = e < E ? ei[E + e] : e - E;
  float a = asrc[s_ * Hh + hd] + adst[d_ * Hh + hd];
  a = a > 0.f ? a : 0.2f * a;
  float ex = expf(a - mx[d_ * Hh + hd]);
  eexp[i] = ex;
  atomicAdd(&den[d_ * Hh + hd], ex);
}

__global__ void k_gat_agg(const int* __restrict__ ei, const float* __restrict__ eexp,
                          const float* __restrict__ den, const float* __restrict__ h,
                          float* __restrict__ out, int E, int Nn, int Hh, int Cc)
{
  int Et = E + Nn;
  long i = blockIdx.x * 256L + threadIdx.x;
  if (i >= (long)Et * Hh * Cc) return;
  int c = i % Cc; long t = i / Cc;
  int hd = t % Hh; int e = (int)(t / Hh);
  int s_ = e < E ? ei[e] : e - E;
  int d_ = e < E ? ei[E + e] : e - E;
  float alpha = eexp[(long)e * Hh + hd] / den[d_ * Hh + hd];
  atomicAdd(&out[((long)d_ * Hh + hd) * Cc + c],
            alpha * h[((long)s_ * Hh + hd) * Cc + c]);
}

__global__ void k_gat_fin(const float* __restrict__ gsum, const float* __restrict__ bias,
                          float* o32, h16* o16, int Nn, int D)
{
  long i = blockIdx.x * 256L + threadIdx.x;
  if (i >= (long)Nn * D) return;
  float v = gsum[i] + bias[i % D];
  v = v > 0.f ? v : expm1f(v);   // ELU
  if (o32) o32[i] = v;
  if (o16) o16[i] = (h16)v;
}

// ---------------------------------------------------------------------------
// Edge features & transformer (scalar parts; <1% of FLOPs)
// ---------------------------------------------------------------------------
__global__ void k_concat(const float* __restrict__ h2, const float* __restrict__ eemb,
                         const int* __restrict__ ei, float* __restrict__ xf,
                         h16* __restrict__ xh, int E)
{
  long i = blockIdx.x * 256L + threadIdx.x;
  if (i >= (long)E * 192) return;
  int c = i % 192; int e = (int)(i / 192);
  float v;
  if (c < 64)       v = h2[(long)ei[e] * 64 + c];
  else if (c < 128) v = h2[(long)ei[E + e] * 64 + (c - 64)];
  else              v = eemb[(long)e * 64 + (c - 128)];
  xf[i] = v; xh[i] = (h16)v;
}

__global__ void k_attn_scores(const float* __restrict__ qkv, float* __restrict__ sc,
                              int E, int Hh, int hd, int d3, float scale)
{
  long total = (long)Hh * E * E;
  long i = blockIdx.x * 256L + threadIdx.x;
  if (i >= total) return;
  int kk = i % E; long t = i / E;
  int q = t % E; int h = (int)(t / E);
  const float* qp = qkv + (long)q * d3 + h * hd;
  const float* kp = qkv + (long)kk * d3 + (d3 / 3) + h * hd;
  float s = 0.f;
  for (int c = 0; c < hd; ++c) s += qp[c] * kp[c];
  sc[i] = s * scale;
}

__global__ void k_softmax_rows(float* __restrict__ sc, int E)
{
  long row = blockIdx.x;
  float* p = sc + row * (long)E;
  __shared__ float red[256];
  int tid = threadIdx.x;
  float m = -3.4e38f;
  for (int i = tid; i < E; i += 256) m = fmaxf(m, p[i]);
  red[tid] = m; __syncthreads();
  for (int s = 128; s > 0; s >>= 1) { if (tid < s) red[tid] = fmaxf(red[tid], red[tid + s]); __syncthreads(); }
  float mx = red[0]; __syncthreads();
  float sum = 0.f;
  for (int i = tid; i < E; i += 256) { float e = expf(p[i] - mx); p[i] = e; sum += e; }
  red[tid] = sum; __syncthreads();
  for (int s = 128; s > 0; s >>= 1) { if (tid < s) red[tid] += red[tid + s]; __syncthreads(); }
  float inv = 1.f / red[0];
  for (int i = tid; i < E; i += 256) p[i] *= inv;
}

__global__ void k_attn_out(const float* __restrict__ sc, const float* __restrict__ qkv,
                           h16* __restrict__ o16, int E, int Hh, int hd, int d3)
{
  long total = (long)E * Hh * hd;
  long i = blockIdx.x * 256L + threadIdx.x;
  if (i >= total) return;
  int c = i % hd; long t = i / hd;
  int h = t % Hh; int q = (int)(t / Hh);
  const float* row = sc + ((long)h * E + q) * E;
  const float* vp = qkv + 2 * (d3 / 3) + h * hd + c;
  float s = 0.f;
  for (int k = 0; k < E; ++k) s += row[k] * vp[(long)k * d3];
  o16[(long)q * (Hh * hd) + h * hd + c] = (h16)s;
}

__global__ void k_res_ln(const float* __restrict__ x, const float* __restrict__ y,
                         const float* __restrict__ g, const float* __restrict__ b,
                         float* __restrict__ xo, h16* __restrict__ xo16, int D)
{
  int row = blockIdx.x; int tid = threadIdx.x;
  __shared__ float red[256];
  float v = 0.f;
  if (tid < D) v = x[(long)row * D + tid] + y[(long)row * D + tid];
  red[tid] = (tid < D) ? v : 0.f; __syncthreads();
  for (int s = 128; s > 0; s >>= 1) { if (tid < s) red[tid] += red[tid + s]; __syncthreads(); }
  float mean = red[0] / D; __syncthreads();
  float c = (tid < D) ? (v - mean) : 0.f;
  red[tid] = c * c; __syncthreads();
  for (int s = 128; s > 0; s >>= 1) { if (tid < s) red[tid] += red[tid + s]; __syncthreads(); }
  float r = rsqrtf(red[0] / D + 1e-5f);
  if (tid < D) {
    float o = g[tid] * (v - mean) * r + b[tid];
    xo[(long)row * D + tid] = o;
    xo16[(long)row * D + tid] = (h16)o;
  }
}

__global__ void k_sigmoid(const float* __restrict__ l, float* __restrict__ o,
                          int E, int Np)
{
  int i = blockIdx.x * 256 + threadIdx.x;
  if (i < E) o[i] = 1.f / (1.f + expf(-l[(long)i * Np]));
}

// ---------------------------------------------------------------------------
// Host orchestration
// ---------------------------------------------------------------------------
static inline dim3 EW(long n) { return dim3((unsigned)((n + 255) / 256)); }

extern "C" void kernel_launch(void* const* d_in, const int* in_sizes, int n_in,
                              void* d_out, int out_size, void* d_ws, size_t ws_size,
                              hipStream_t stream)
{
  (void)in_sizes; (void)n_in; (void)out_size;
  const float* x_in = (const float*)d_in[0];
  const int*   ei   = (const int*)d_in[1];
  const float* ea   = (const float*)d_in[2];

  int p = 3;
  auto F = [&]() { return (const float*)d_in[p++]; };
  const float *Wf1=F(),*Wf2=F(),*Wo=F(),*Wqkv=F(),*bf1=F(),*bf2=F(),*bo=F(),*bqkv=F();
  const float *ln1b=F(),*ln1g=F(),*ln2b=F(),*ln2g=F();
  const float *eW1=F(),*eW2=F(),*eb1=F(),*eb2=F();
  const float *g1W=F(),*g1ad=F(),*g1as=F(),*g1b=F();
  const float *g2W=F(),*g2ad=F(),*g2as=F(),*g2b=F();
  const float *mW1=F(),*mW2=F(),*mb1=F(),*mb2=F();
  const float *bn1b=F(),*bn1g=F(),*bn1m=F(),*bn1v=F();
  const float *cw1=F();
  struct Blk { const float *b1b,*b1g,*b1m,*b1v,*b2b,*b2g,*b2m,*b2v,*w1,*w2,
                           *dbb,*dbg,*dbm,*dbv,*dw; bool down; };
  Blk blks[16]; int nbt = 0;
  const int nblk[4] = {3, 4, 6, 3};
  for (int li = 0; li < 4; ++li)
    for (int bi = 0; bi < nblk[li]; ++bi) {
      Blk b{};
      b.b1b=F(); b.b1g=F(); b.b1m=F(); b.b1v=F();
      b.b2b=F(); b.b2g=F(); b.b2m=F(); b.b2v=F();
      b.w1=F();  b.w2=F();
      b.down = (li > 0 && bi == 0);
      if (b.down) { b.dbb=F(); b.dbg=F(); b.dbm=F(); b.dbv=F(); b.dw=F(); }
      blks[nbt++] = b;
    }

  uint8_t* ws = (uint8_t*)d_ws; size_t off = 0;
  auto AL = [&](size_t bytes) -> void* {
    void* r = ws + off; off = (off + bytes + 255) & ~(size_t)255; return r;
  };
  h16*  nhwc  = (h16*) AL(256L*64*64*3*2);
  h16*  B0    = (h16*) AL(256L*32*32*64*2);
  h16*  B1    = (h16*) AL(256L*32*32*64*2);
  h16*  B2    = (h16*) AL(256L*32*32*64*2);
  h16*  imb   = (h16*) AL(262144L*160*2);
  h16*  wb    = (h16*) AL(4608L*512*2);
  float* bbuf = (float*)AL(576*4);
  h16*  apk   = (h16*) AL(1024L*256*2);    // packed-A scratch for small GEMMs
  h16*  feats = (h16*) AL(256L*512*2);
  float* hpre = (float*)AL(256L*256*4);
  float* asrc = (float*)AL(256*4*4);
  float* adst = (float*)AL(256*4*4);
  float* mxb  = (float*)AL(256*4*4);
  float* den  = (float*)AL(256*4*4);
  float* eexp = (float*)AL(1280L*4*4);
  float* gout = (float*)AL(256L*256*4);
  h16*  h1h   = (h16*) AL(256L*256*2);
  float* h2   = (float*)AL(256L*64*4);
  h16*  ea16  = (h16*) AL(1024L*32*2);
  h16*  e1h   = (h16*) AL(1024L*64*2);
  float* eemb = (float*)AL(1024L*64*4);
  float* xf   = (float*)AL(1024L*192*4);
  h16*  xh    = (h16*) AL(1024L*192*2);
  float* qkv  = (float*)AL(1024L*576*4);
  float* sc   = (float*)AL(4L*1024*1024*4);
  h16*  o16   = (h16*) AL(1024L*192*2);
  float* tmpf = (float*)AL(1024L*192*4);
  h16*  ffh   = (h16*) AL(1024L*256*2);
  h16*  m1h   = (h16*) AL(1024L*64*2);
  float* logit= (float*)AL(1024L*64*4);
  if (off > ws_size) return;

  auto GEMM = [&](const h16* A, const h16* Bt, const float* bias, const h16* res,
                  h16* Ch, float* Cf, int M, int N, int K, int relu) {
    dim3 g(N / 64, M / 64);
    k_gemm<<<g, dim3(128), 0, stream>>>(A, Bt, bias, res, Ch, Cf, M, N, K, relu);
  };
  // pack a row-major f16 [M,K] into fragment layout, then GEMM from apk
  auto PACK = [&](const h16* src, int M, int K) {
    k_pack_a<<<EW((long)M * K), 256, 0, stream>>>(src, apk, M, K);
  };

  auto CONV = [&](const h16* src, int H, int W, int Ci, int KH, int KW, int s,
                  int pad, const float* w, const float* bg, const float* bbeta,
                  const float* bm, const float* bv, const h16* res, int relu,
                  h16* out, int Co) {
    int OH = (H + 2 * pad - KH) / s + 1;
    int OW = (W + 2 * pad - KW) / s + 1;
    int Kreal = Ci * KH * KW;
    int Kp = (Kreal + 31) & ~31;
    long M = 256L * OH * OW;
    k_im2col<<<EW(M * Kp), 256, 0, stream>>>(src, imb, 256, H, W, Ci, KH, KW,
                                             s, pad, OH, OW, Kreal, Kp);
    k_prep_wconv<<<EW((long)Kp * Co), 256, 0, stream>>>(w, bg, bv, wb, Co, Ci, KH, KW, Kp);
    k_bn_bias<<<EW(Co), 256, 0, stream>>>(bg, bbeta, bm, bv, bbuf, Co);
    GEMM(imb, wb, bbuf, res, out, nullptr, (int)M, Co, Kp, relu);
  };

  // ================= ResNet34 =================
  k_nchw_to_nhwc<<<EW(256L*3*64*64), 256, 0, stream>>>(x_in, nhwc, 256, 3, 64, 64);
  CONV(nhwc, 64, 64, 3, 7, 7, 2, 3, cw1, bn1g, bn1b, bn1m, bn1v, nullptr, 1, B0, 64);
  k_maxpool<<<EW(256L*16*16*64), 256, 0, stream>>>(B0, B1, 256, 32, 32, 64, 16, 16);

  h16 *X = B1, *Fa = B0, *Fb = B2;
  int H = 16, Wd = 16, Ci = 64;
  const int couts[4] = {64, 128, 256, 512};
  int bix = 0;
  for (int li = 0; li < 4; ++li)
    for (int bi = 0; bi < nblk[li]; ++bi) {
      Blk& b = blks[bix++];
      int Co = couts[li];
      int s = (li > 0 && bi == 0) ? 2 : 1;
      int OH = (H - 1) / s + 1, OW = (Wd - 1) / s + 1;
      CONV(X, H, Wd, Ci, 3, 3, s, 1, b.w1, b.b1g, b.b1b, b.b1m, b.b1v,
           nullptr, 1, Fa, Co);
      const h16* scp;
      if (b.down) {
        CONV(X, H, Wd, Ci, 1, 1, s, 0, b.dw, b.dbg, b.dbb, b.dbm, b.dbv,
             nullptr, 0, Fb, Co);
        scp = Fb;
      } else scp = X;
      CONV(Fa, OH, OW, Co, 3, 3, 1, 1, b.w2, b.b2g, b.b2b, b.b2m, b.b2v,
           scp, 1, Fa, Co);
      h16* ox = X; X = Fa; Fa = ox;
      H = OH; Wd = OW; Ci = Co;
    }
  k_avgpool<<<EW(256L*512), 256, 0, stream>>>(X, feats, 256, 2, 2, 512);

  // ================= GAT x2 =================
  auto GAT = [&](const h16* hin, int Kin, const float* W, const float* as_,
                 const float* ad_, const float* bias, int Hh, int Cc,
                 h16* out16, float* out32) {
    int D = Hh * Cc;
    k_prep_lin<<<EW((long)Kin * D), 256, 0, stream>>>(W, wb, Kin, D, Kin, D);
    PACK(hin, 256, Kin);
    GEMM(apk, wb, nullptr, nullptr, nullptr, hpre, 256, D, Kin, 0);
    k_gat_scores<<<EW(256L * Hh), 256, 0, stream>>>(hpre, as_, ad_, asrc, adst, 256, Hh, Cc);
    k_fill<<<EW(256L * Hh), 256, 0, stream>>>(mxb, -3.4e38f, 256L * Hh);
    k_fill<<<EW(256L * Hh), 256, 0, stream>>>(den, 0.f, 256L * Hh);
    k_fill<<<EW(256L * D), 256, 0, stream>>>(gout, 0.f, 256L * D);
    long Et = 1024 + 256;
    k_gat_max<<<EW(Et * Hh), 256, 0, stream>>>(ei, asrc, adst, mxb, 1024, 256, Hh);
    k_gat_exp<<<EW(Et * Hh), 256, 0, stream>>>(ei, asrc, adst, mxb, den, eexp, 1024, 256, Hh);
    k_gat_agg<<<EW(Et * Hh * Cc), 256, 0, stream>>>(ei, eexp, den, hpre, gout, 1024, 256, Hh, Cc);
    k_gat_fin<<<EW(256L * D), 256, 0, stream>>>(gout, bias, out32, out16, 256, D);
  };
  GAT(feats, 512, g1W, g1as, g1ad, g1b, 4, 64, h1h, nullptr);
  GAT(h1h,   256, g2W, g2as, g2ad, g2b, 1, 64, nullptr, h2);

  // ================= edge MLP =================
  k_pad_a16<<<EW(1024L*32), 256, 0, stream>>>(ea, ea16, 1024, 16, 32);
  k_prep_lin<<<EW(32L*64), 256, 0, stream>>>(eW1, wb, 16, 64, 32, 64);
  k_pad_bias<<<EW(64), 256, 0, stream>>>(eb1, bbuf, 64, 64);
  PACK(ea16, 1024, 32);
  GEMM(apk, wb, bbuf, nullptr, e1h, nullptr, 1024, 64, 32, 1);
  k_prep_lin<<<EW(64L*64), 256, 0, stream>>>(eW2, wb, 64, 64, 64, 64);
  k_pad_bias<<<EW(64), 256, 0, stream>>>(eb2, bbuf, 64, 64);
  PACK(e1h, 1024, 64);
  GEMM(apk, wb, bbuf, nullptr, nullptr, eemb, 1024, 64, 64, 0);

  k_concat<<<EW(1024L*192), 256, 0, stream>>>(h2, eemb, ei, xf, xh, 1024);

  // ================= transformer layer =================
  k_prep_lin<<<EW(192L*576), 256, 0, stream>>>(Wqkv, wb, 192, 576, 192, 576);
  k_pad_bias<<<EW(576), 256, 0, stream>>>(bqkv, bbuf, 576, 576);
  PACK(xh, 1024, 192);
  GEMM(apk, wb, bbuf, nullptr, nullptr, qkv, 1024, 576, 192, 0);
  k_attn_scores<<<EW(4L*1024*1024), 256, 0, stream>>>(qkv, sc, 1024, 4, 48, 576,
                                                      1.f / sqrtf(48.f));
  k_softmax_rows<<<dim3(4096), dim3(256), 0, stream>>>(sc, 1024);
  k_attn_out<<<EW(1024L*192), 256, 0, stream>>>(sc, qkv, o16, 1024, 4, 48, 576);
  k_prep_lin<<<EW(192L*192), 256, 0, stream>>>(Wo, wb, 192, 192, 192, 192);
  k_pad_bias<<<EW(192), 256, 0, stream>>>(bo, bbuf, 192, 192);
  PACK(o16, 1024, 192);
  GEMM(apk, wb, bbuf, nullptr, nullptr, tmpf, 1024, 192, 192, 0);
  k_res_ln<<<dim3(1024), dim3(256), 0, stream>>>(xf, tmpf, ln1g, ln1b, xf, xh, 192);
  k_prep_lin<<<EW(192L*256), 256, 0, stream>>>(Wf1, wb, 192, 256, 192, 256);
  k_pad_bias<<<EW(256), 256, 0, stream>>>(bf1, bbuf, 256, 256);
  PACK(xh, 1024, 192);
  GEMM(apk, wb, bbuf, nullptr, ffh, nullptr, 1024, 256, 192, 1);
  k_prep_lin<<<EW(256L*192), 256, 0, stream>>>(Wf2, wb, 256, 192, 256, 192);
  k_pad_bias<<<EW(192), 256, 0, stream>>>(bf2, bbuf, 192, 192);
  PACK(ffh, 1024, 256);
  GEMM(apk, wb, bbuf, nullptr, nullptr, tmpf, 1024, 192, 256, 0);
  k_res_ln<<<dim3(1024), dim3(256), 0, stream>>>(xf, tmpf, ln2g, ln2b, xf, xh, 192);

  // ================= head MLP + sigmoid =================
  k_prep_lin<<<EW(192L*64), 256, 0, stream>>>(mW1, wb, 192, 64, 192, 64);
  k_pad_bias<<<EW(64), 256, 0, stream>>>(mb1, bbuf, 64, 64);
  PACK(xh, 1024, 192);
  GEMM(apk, wb, bbuf, nullptr, m1h, nullptr, 1024, 64, 192, 1);
  k_prep_lin<<<EW(64L*64), 256, 0, stream>>>(mW2, wb, 64, 1, 64, 64);
  k_pad_bias<<<EW(64), 256, 0, stream>>>(mb2, bbuf, 1, 64);
  PACK(m1h, 1024, 64);
  GEMM(apk, wb, bbuf, nullptr, nullptr, logit, 1024, 64, 64, 0);
  k_sigmoid<<<EW(1024), 256, 0, stream>>>(logit, (float*)d_out, 1024, 64);
}
